// TemporalRoPEAttention_1408749273522
// MI455X (gfx1250) — matrix operand order repaired and checked by reading, not compiled, not run on hardware
//
#include <hip/hip_runtime.h>
#include <hip/hip_bf16.h>

// ---------------- problem constants ----------------
#define B_     4
#define NTOK   1537          // V*T + 1
#define VT_    1536          // V*T
#define V_     12
#define DIM_   1024
#define HEADS_ 16
#define HD_    64
#define TD_    3072          // 3*DIM
#define MTOT   (B_ * NTOK)   // 6148 rows of (b,n)

typedef _Float16 v16h __attribute__((ext_vector_type(16)));
typedef _Float16 v8h  __attribute__((ext_vector_type(8)));
typedef float    v8f  __attribute__((ext_vector_type(8)));

static __device__ inline v8f wmma_f16(v16h a, v16h b, v8f c) {
    // D = A(16x32) * B(32x16) + C, f32 accumulate
    return __builtin_amdgcn_wmma_f32_16x16x32_f16(
        /*neg_a=*/false, a, /*neg_b=*/false, b,
        /*c_mod=*/(short)0, c, /*reuse_a=*/false, /*reuse_b=*/false);
}

// ---------------- f32 -> f16 convert ----------------
__global__ void conv_kernel(const float* __restrict__ src, _Float16* __restrict__ dst, long n) {
    long i = (long)blockIdx.x * blockDim.x + threadIdx.x;
    if (i < n) dst[i] = (_Float16)src[i];
}

// ---------------- argsort rank -> RoPE position table ----------------
// pos[b*NTOK + 0] = 0 (CLS, no rotation); pos[b*NTOK + 1 + rank(j)] = (j % V) + 1
__global__ void pos_kernel(const int* __restrict__ ids, int* __restrict__ pos) {
    int tid = blockIdx.x * blockDim.x + threadIdx.x;
    if (tid < B_ * VT_) {
        int b = tid / VT_, j = tid % VT_;
        const int* row = ids + b * VT_;
        int val = row[j];
        int rank = 0;
        for (int i = 0; i < VT_; ++i) {
            int vi = row[i];
            rank += (vi < val) || (vi == val && i < j);   // stable argsort tie-break
        }
        pos[b * NTOK + 1 + rank] = (j % V_) + 1;
    } else if (tid < B_ * VT_ + B_) {
        pos[(tid - B_ * VT_) * NTOK] = 0;
    }
}

// ---------------- RoPE rotate (deinterleave) + v convert ----------------
// qkv: f32 [MTOT, 3072]. Outputs f16 [MTOT, 1024] each.
__global__ void rope_kernel(const float* __restrict__ qkv, const int* __restrict__ pos,
                            _Float16* __restrict__ qr, _Float16* __restrict__ kr,
                            _Float16* __restrict__ vh) {
    long tid = (long)blockIdx.x * blockDim.x + threadIdx.x;
    if (tid >= (long)MTOT * (HEADS_ * 32)) return;
    int i  = (int)(tid & 31);          // pair index 0..31
    int hh = (int)((tid >> 5) & 15);   // head
    int m  = (int)(tid >> 9);          // flat (b,n) row
    int p  = pos[m];
    const float* src = qkv + (size_t)m * TD_ + hh * HD_;
    float reQ = src[2 * i],        imQ = src[2 * i + 1];
    float reK = src[1024 + 2 * i], imK = src[1024 + 2 * i + 1];
    float c = 1.0f, s = 0.0f;
    if (p > 0) {
        float theta = __powf(10000.0f, -(float)(2 * i) * (1.0f / 64.0f));
        float ang = (float)p * theta;
        __sincosf(ang, &s, &c);
    }
    _Float16* qd = qr + (size_t)m * DIM_ + hh * HD_;
    qd[i]      = (_Float16)(reQ * c - imQ * s);
    qd[32 + i] = (_Float16)(reQ * s + imQ * c);
    _Float16* kd = kr + (size_t)m * DIM_ + hh * HD_;
    kd[i]      = (_Float16)(reK * c - imK * s);
    kd[32 + i] = (_Float16)(reK * s + imK * c);
    _Float16* vd = vh + (size_t)m * DIM_ + hh * HD_;
    vd[2 * i]     = (_Float16)src[2048 + 2 * i];
    vd[2 * i + 1] = (_Float16)src[2048 + 2 * i + 1];
}

// ---------------- WMMA GEMM: C[M,N] = A[M,K] * W[N,K]^T (+ bias) ----------------
// block = 256 threads = 8 waves arranged 4(M) x 2(N); wave tile 32x64 (8 wmma / K-step);
// block tile 128(M) x 128(N). Nout must be a multiple of 128.
__global__ __launch_bounds__(256) void gemm_f16(
    const _Float16* __restrict__ A, const _Float16* __restrict__ W,
    const float* __restrict__ bias, float* __restrict__ Cf, _Float16* __restrict__ Ch,
    int M, int Nout, int K, float hscale)
{
    int wid  = threadIdx.x >> 5;
    int lane = threadIdx.x & 31;
    bool hiHalf = lane >= 16;
    int  l15 = lane & 15;
    int wm = wid & 3, wn = wid >> 2;
    int m_base = blockIdx.x * 128 + wm * 32;
    int n_base = blockIdx.y * 128 + wn * 64;

    // clamp rows (padding rows read row M-1; their results are never stored)
    int rowA[2];
    #pragma unroll
    for (int t = 0; t < 2; ++t) {
        int r = m_base + t * 16 + l15;
        rowA[t] = r < M ? r : M - 1;
    }

    v8f acc[2][4] = {};
    for (int k0 = 0; k0 < K; k0 += 32) {
        if (k0 + 32 < K) {  // prefetch next K-tile (global_prefetch on gfx1250)
            __builtin_prefetch(A + (size_t)rowA[0] * K + k0 + 32, 0, 1);
            __builtin_prefetch(W + (size_t)(n_base + l15) * K + k0 + 32, 0, 1);
        }
        v16h af[2], bf[4];
        #pragma unroll
        for (int t = 0; t < 2; ++t) {            // A fragments (clamped rows)
            const _Float16* p = A + (size_t)rowA[t] * K + k0 + (hiHalf ? 8 : 0);
            v16h a;
            #pragma unroll
            for (int q = 0; q < 8; ++q) { a[q] = p[q]; a[8 + q] = p[16 + q]; }
            af[t] = a;
        }
        #pragma unroll
        for (int t = 0; t < 4; ++t) {            // B fragments: weight rows = output cols
            int col = n_base + t * 16 + l15;
            const _Float16* p = W + (size_t)col * K + k0 + (hiHalf ? 8 : 0);
            v16h bv;
            #pragma unroll
            for (int q = 0; q < 8; ++q) { bv[q] = p[q]; bv[8 + q] = p[16 + q]; }
            bf[t] = bv;
        }
        #pragma unroll
        for (int tm = 0; tm < 2; ++tm)
            #pragma unroll
            for (int tn = 0; tn < 4; ++tn)
                acc[tm][tn] = wmma_f16(af[tm], bf[tn], acc[tm][tn]);
    }
    // epilogue: C layout -> lanes 0-15: M=j ; lanes 16-31: M=8+j ; N = l15
    #pragma unroll
    for (int tm = 0; tm < 2; ++tm)
        #pragma unroll
        for (int tn = 0; tn < 4; ++tn)
            #pragma unroll
            for (int j = 0; j < 8; ++j) {
                int row = m_base + tm * 16 + (hiHalf ? 8 + j : j);
                int col = n_base + tn * 16 + l15;
                if (row < M) {
                    float v = acc[tm][tn][j] + (bias ? bias[col] : 0.0f);
                    if (Cf) Cf[(size_t)row * Nout + col] = v;
                    if (Ch) Ch[(size_t)row * Nout + col] = (_Float16)(v * hscale);
                }
            }
}

// ---------------- flash attention ----------------
// grid.x = B*HEADS*ceil(N/64); block = 128 (4 waves); wave = 16 query rows.
// All 4 waves share (b,h): K/V 32x64 tiles are staged cooperatively in LDS.
// Q is pre-scaled by 1/sqrt(HD).
__global__ __launch_bounds__(128) void attn_kernel(
    const _Float16* __restrict__ Q, const _Float16* __restrict__ Kh,
    const _Float16* __restrict__ Vh, _Float16* __restrict__ ctx)
{
    const int QB = (NTOK + 63) / 64;
    int qb = blockIdx.x % QB;
    int bh = blockIdx.x / QB;
    int h = bh % HEADS_, b = bh / HEADS_;
    int wid  = threadIdx.x >> 5;
    int lane = threadIdx.x & 31;
    bool hiHalf = lane >= 16;
    int  l15 = lane & 15;
    int q0 = qb * 64 + wid * 16;
    const size_t base = (size_t)b * NTOK * DIM_ + (size_t)h * HD_;

    __shared__ _Float16 kbuf[32 * 64];
    __shared__ _Float16 vbuf[32 * 64];
    __shared__ float    pbuf[4][16 * 32];

    // Q fragments (16 x 64 = 2 K-slabs of 32), row clamped (stores are guarded)
    int qrow = q0 + l15; qrow = qrow < NTOK ? qrow : NTOK - 1;
    v16h qf[2];
    #pragma unroll
    for (int kw = 0; kw < 2; ++kw) {
        const _Float16* p = Q + base + (size_t)qrow * DIM_ + kw * 32 + (hiHalf ? 8 : 0);
        v16h a;
        #pragma unroll
        for (int q = 0; q < 8; ++q) { a[q] = p[q]; a[8 + q] = p[16 + q]; }
        qf[kw] = a;
    }

    v8f o[4] = {};
    float mrow[8], lrow[8];
    #pragma unroll
    for (int j = 0; j < 8; ++j) { mrow[j] = -1e30f; lrow[j] = 0.0f; }

    // cooperative staging coordinates: 128 threads x 16 f16 = 32x64 tile
    int sr = threadIdx.x >> 2;           // tile row 0..31
    int se = (threadIdx.x & 3) * 16;     // element offset 0/16/32/48

    const int nkt = (NTOK + 31) / 32;
    for (int kt = 0; kt < nkt; ++kt) {
        // --- stage K and V tiles into LDS ---
        __syncthreads();                  // previous iteration's readers are done
        {
            int tok = kt * 32 + sr;
            int tc = tok < NTOK ? tok : NTOK - 1;
            const _Float16* gk = Kh + base + (size_t)tc * DIM_ + se;
            const _Float16* gv = Vh + base + (size_t)tc * DIM_ + se;
            #pragma unroll
            for (int q = 0; q < 16; ++q) {
                kbuf[sr * 64 + se + q] = gk[q];
                vbuf[sr * 64 + se + q] = gv[q];
            }
        }
        __syncthreads();                  // tiles visible to all waves

        // --- S = Q * K^T : two 16x16 tiles over 32 tokens (K from LDS) ---
        v8f s[2];
        #pragma unroll
        for (int sb = 0; sb < 2; ++sb) {
            int tok = kt * 32 + sb * 16 + l15;
            v8f sc = {};
            #pragma unroll
            for (int kw = 0; kw < 2; ++kw) {
                const _Float16* p = &kbuf[(sb * 16 + l15) * 64 + kw * 32 + (hiHalf ? 8 : 0)];
                v16h bv;
                #pragma unroll
                for (int q = 0; q < 8; ++q) { bv[q] = p[q]; bv[8 + q] = p[16 + q]; }
                sc = wmma_f16(qf[kw], bv, sc);
            }
            if (tok >= NTOK) {
                #pragma unroll
                for (int j = 0; j < 8; ++j) sc[j] = -1e30f;   // mask padded tokens
            }
            s[sb] = sc;
        }
        // --- online softmax (row stats replicated across 16-lane halves) ---
        float alpha[8];
        #pragma unroll
        for (int j = 0; j < 8; ++j) {
            float v = fmaxf(s[0][j], s[1][j]);
            #pragma unroll
            for (int off = 1; off < 16; off <<= 1) v = fmaxf(v, __shfl_xor(v, off, 32));
            float mnew = fmaxf(mrow[j], v);
            alpha[j] = __expf(mrow[j] - mnew);
            mrow[j]  = mnew;
            float p0 = __expf(s[0][j] - mnew);
            float p1 = __expf(s[1][j] - mnew);
            s[0][j] = p0; s[1][j] = p1;
            float rs = p0 + p1;
            #pragma unroll
            for (int off = 1; off < 16; off <<= 1) rs += __shfl_xor(rs, off, 32);
            lrow[j] = lrow[j] * alpha[j] + rs;
        }
        #pragma unroll
        for (int nt = 0; nt < 4; ++nt)
            #pragma unroll
            for (int j = 0; j < 8; ++j) o[nt][j] *= alpha[j];

        // --- re-layout P (C fragment -> A fragment) via per-wave LDS region ---
        // (no barrier needed: same-wave LDS ops are in-order on CDNA5)
        #pragma unroll
        for (int j = 0; j < 8; ++j) {
            int row = hiHalf ? 8 + j : j;
            pbuf[wid][row * 32 + l15]      = s[0][j];
            pbuf[wid][row * 32 + 16 + l15] = s[1][j];
        }
        v16h pf;
        {
            const float* pp = &pbuf[wid][l15 * 32 + (hiHalf ? 8 : 0)];
            #pragma unroll
            for (int q = 0; q < 8; ++q) {
                pf[q]     = (_Float16)pp[q];
                pf[8 + q] = (_Float16)pp[16 + q];
            }
        }
        // --- O += P * V  (V fragments from LDS; 4 n-tiles cover head_dim 64) ---
        #pragma unroll
        for (int nt = 0; nt < 4; ++nt) {
            v16h vf;
            #pragma unroll
            for (int kk = 0; kk < 16; ++kk)
                vf[kk] = vbuf[((hiHalf ? 8 : 0) + kk + (kk >= 8 ? 8 : 0)) * 64 + nt * 16 + l15];
            o[nt] = wmma_f16(pf, vf, o[nt]);
        }
    }

    // --- normalize and store ctx (f16 for next GEMM) ---
    float invl[8];
    #pragma unroll
    for (int j = 0; j < 8; ++j) invl[j] = 1.0f / lrow[j];
    #pragma unroll
    for (int nt = 0; nt < 4; ++nt)
        #pragma unroll
        for (int j = 0; j < 8; ++j) {
            int row = q0 + (hiHalf ? 8 + j : j);
            if (row < NTOK)
                ctx[base + (size_t)row * DIM_ + nt * 16 + l15] = (_Float16)(o[nt][j] * invl[j]);
        }
}

// ---------------- host launcher ----------------
extern "C" void kernel_launch(void* const* d_in, const int* in_sizes, int n_in,
                              void* d_out, int out_size, void* d_ws, size_t ws_size,
                              hipStream_t stream) {
    const float* x      = (const float*)d_in[0];
    const float* qkv_w  = (const float*)d_in[1];
    const float* in_w   = (const float*)d_in[2];
    const float* in_b   = (const float*)d_in[3];
    const float* out_w  = (const float*)d_in[4];
    const float* out_b  = (const float*)d_in[5];
    const float* proj_w = (const float*)d_in[6];
    const float* proj_b = (const float*)d_in[7];
    const int*   ids    = (const int*)d_in[8];
    float* out = (float*)d_out;

    char* ws = (char*)d_ws;
    size_t off = 0;
    auto alloc = [&](size_t bytes) -> void* {
        off = (off + 255) & ~(size_t)255;
        void* p = ws + off;
        off += bytes;
        return p;
    };

    _Float16* x_h    = (_Float16*)alloc((size_t)MTOT * DIM_ * 2);
    _Float16* qkvw_h = (_Float16*)alloc((size_t)TD_ * DIM_ * 2);
    _Float16* inw_h  = (_Float16*)alloc((size_t)TD_ * DIM_ * 2);
    _Float16* ow_h   = (_Float16*)alloc((size_t)DIM_ * DIM_ * 2);
    _Float16* pw_h   = (_Float16*)alloc((size_t)DIM_ * DIM_ * 2);
    float*    qkv_f  = (float*)   alloc((size_t)MTOT * TD_ * 4);
    _Float16* qr_h   = (_Float16*)alloc((size_t)MTOT * DIM_ * 2);
    _Float16* kr_h   = (_Float16*)alloc((size_t)MTOT * DIM_ * 2);
    _Float16* v_h    = (_Float16*)alloc((size_t)MTOT * DIM_ * 2);
    _Float16* qp_h   = (_Float16*)alloc((size_t)MTOT * DIM_ * 2);
    _Float16* kp_h   = (_Float16*)alloc((size_t)MTOT * DIM_ * 2);
    _Float16* vp_h   = (_Float16*)alloc((size_t)MTOT * DIM_ * 2);
    _Float16* ctx_h  = (_Float16*)alloc((size_t)MTOT * DIM_ * 2);
    _Float16* mha_h  = (_Float16*)alloc((size_t)MTOT * DIM_ * 2);
    int*      pos    = (int*)     alloc((size_t)B_ * NTOK * 4);
    (void)ws_size; (void)in_sizes; (void)n_in; (void)out_size;

    auto cdiv = [](long a, long b) { return (a + b - 1) / b; };

    // 1) f32 -> f16 conversions
    {
        long n;
        n = (long)MTOT * DIM_; conv_kernel<<<cdiv(n, 256), 256, 0, stream>>>(x, x_h, n);
        n = (long)TD_ * DIM_;  conv_kernel<<<cdiv(n, 256), 256, 0, stream>>>(qkv_w, qkvw_h, n);
        n = (long)TD_ * DIM_;  conv_kernel<<<cdiv(n, 256), 256, 0, stream>>>(in_w, inw_h, n);
        n = (long)DIM_ * DIM_; conv_kernel<<<cdiv(n, 256), 256, 0, stream>>>(out_w, ow_h, n);
        n = (long)DIM_ * DIM_; conv_kernel<<<cdiv(n, 256), 256, 0, stream>>>(proj_w, pw_h, n);
    }
    // 2) RoPE position table (rank-based stable argsort)
    pos_kernel<<<cdiv(B_ * VT_ + B_, 256), 256, 0, stream>>>(ids, pos);

    // 3) qkv = x @ qkv_w^T  (f32 out)
    {
        dim3 grid((unsigned)cdiv(MTOT, 128), TD_ / 128);
        gemm_f16<<<grid, 256, 0, stream>>>(x_h, qkvw_h, nullptr, qkv_f, nullptr,
                                           MTOT, TD_, DIM_, 1.0f);
    }
    // 4) RoPE rotate q,k + convert v
    {
        long n = (long)MTOT * HEADS_ * 32;
        rope_kernel<<<cdiv(n, 256), 256, 0, stream>>>(qkv_f, pos, qr_h, kr_h, v_h);
    }
    // 5) in_proj: qp (pre-scaled by 1/8), kp, vp  (f16 out)
    {
        dim3 grid((unsigned)cdiv(MTOT, 128), DIM_ / 128);
        gemm_f16<<<grid, 256, 0, stream>>>(qr_h, inw_h, in_b, nullptr, qp_h,
                                           MTOT, DIM_, DIM_, 0.125f);
        gemm_f16<<<grid, 256, 0, stream>>>(kr_h, inw_h + (size_t)DIM_ * DIM_, in_b + DIM_,
                                           nullptr, kp_h, MTOT, DIM_, DIM_, 1.0f);
        gemm_f16<<<grid, 256, 0, stream>>>(v_h, inw_h + (size_t)2 * DIM_ * DIM_, in_b + 2 * DIM_,
                                           nullptr, vp_h, MTOT, DIM_, DIM_, 1.0f);
    }
    // 6) flash attention -> ctx (f16)
    {
        int QB = (NTOK + 63) / 64;
        attn_kernel<<<B_ * HEADS_ * QB, 128, 0, stream>>>(qp_h, kp_h, vp_h, ctx_h);
    }
    // 7) out_proj -> mha (f16), then proj -> out (f32)
    {
        dim3 grid((unsigned)cdiv(MTOT, 128), DIM_ / 128);
        gemm_f16<<<grid, 256, 0, stream>>>(ctx_h, ow_h, out_b, nullptr, mha_h,
                                           MTOT, DIM_, DIM_, 1.0f);
        gemm_f16<<<grid, 256, 0, stream>>>(mha_h, pw_h, proj_b, out, nullptr,
                                           MTOT, DIM_, DIM_, 1.0f);
    }
}
